// Attention_12824772346342
// MI455X (gfx1250) — matrix operand order repaired and checked by reading, not compiled, not run on hardware
//
#include <hip/hip_runtime.h>

// Causal attention, B=2 H=16 S=2048 D=64, fp32 in/out, bf16 WMMA compute.
// Flash-attention: each workgroup = 128 query rows of one (b,h); 8 waves,
// wave w owns 16 query COLUMNS of the transposed-score formulation.
//   S^T = K * Q^T      (A = K rows from LDS, B = Q^T rows in registers)
//   O^T += V^T * P^T   (A = V^T rows from LDS, B = P^T packed in registers)
// K/V tiles are double-buffered in LDS: one barrier per key block, and the
// next tile's load+bf16-convert overlaps the current tile's WMMA work.

#define B_ 2
#define H_ 16
#define S_ 2048
#define D_ 64

#define KV_STRIDE 72          // bf16 row stride (bank-conflict pad)
#define KV_TILE   (64 * KV_STRIDE)

typedef __attribute__((ext_vector_type(16))) __bf16 v16bf;
typedef __attribute__((ext_vector_type(8)))  float  v8f;

union BF16x16 {
  v16bf v;
  unsigned int u[8];
  uint4 q[2];
};

// ---- fp32 pair -> packed bf16 dword ----
// round-half-up bias then one v_perm_b32 picking the two high halves.
__device__ __forceinline__ unsigned int packbf(float lo, float hi) {
  unsigned int a = __float_as_uint(lo) + 0x8000u;
  unsigned int b = __float_as_uint(hi) + 0x8000u;
#if __has_builtin(__builtin_amdgcn_perm)
  return __builtin_amdgcn_perm(b, a, 0x07060302u);  // {b[31:16], a[31:16]}
#else
  return (a >> 16) | (b & 0xffff0000u);
#endif
}

// ---- raw v_exp_f32 ----
__device__ __forceinline__ float fast_exp2(float x) {
#if __has_builtin(__builtin_amdgcn_exp2f)
  return __builtin_amdgcn_exp2f(x);
#else
  return exp2f(x);
#endif
}

__device__ __forceinline__ float fast_rcp(float x) {
#if __has_builtin(__builtin_amdgcn_rcpf)
  return __builtin_amdgcn_rcpf(x);
#else
  return 1.0f / x;
#endif
}

// ---- half-wave swap: res.x = {a[0:15], b[0:15]}, res.y = {a[16:31], b[16:31]} ----
struct U2 { unsigned int x, y; };
__device__ __forceinline__ U2 halfswap(unsigned int a, unsigned int b) {
  U2 r;
#if __has_builtin(__builtin_amdgcn_permlane16_swap)
  auto t = __builtin_amdgcn_permlane16_swap(a, b, false, false);
  r.x = (unsigned int)t[0];
  r.y = (unsigned int)t[1];
#else
  unsigned int xa = (unsigned int)__shfl_xor((int)a, 16, 32);
  unsigned int xb = (unsigned int)__shfl_xor((int)b, 16, 32);
  int sel = (threadIdx.x & 31) >> 4;
  r.x = sel ? xb : a;
  r.y = sel ? b : xa;
#endif
  return r;
}

__global__ __launch_bounds__(256)
void fa_cdna5_kernel(const float* __restrict__ Q, const float* __restrict__ K,
                     const float* __restrict__ V, float* __restrict__ O) {
  // double-buffered K tile: [buf][key 0..63][d 0..63] bf16
  __shared__ __align__(16) unsigned short sK[2 * KV_TILE];
  // double-buffered V tile transposed: [buf][d 0..63][key 0..63] bf16
  __shared__ __align__(16) unsigned short sVt[2 * KV_TILE];
  // per-wave output staging for coalesced stores: [wave][q 0..15][d 0..63]+pad
  __shared__ __align__(16) float sO[8 * 16 * 68];

  const int tid  = threadIdx.x;
  const int w    = tid >> 5;       // wave id 0..7
  const int lane = tid & 31;
  const int sel  = lane >> 4;      // half-wave
  const int qc   = lane & 15;      // q column / matrix row within tile

  const int q0   = blockIdx.x * 128;   // 128 query rows per workgroup
  const int jmax = 2 * blockIdx.x + 1; // key blocks 0..jmax (64 keys each)
  const int bh   = blockIdx.y;
  const size_t base = (size_t)bh * S_ * D_;
  const float* Qg = Q + base;
  const float* Kg = K + base;
  const float* Vg = V + base;
  float*       Og = O + base;

  const int qg = q0 + w * 16 + qc;     // this lane's global query index

  // per-thread fill coordinates (constant across iterations)
  const int frow = tid >> 4, fseg = tid & 15;   // K fill: row=key, seg=d/4
  const int frp  = tid >> 4;                    // V fill: key pair

  // ---- stage one 64-key K/V tile (bf16 convert, V transposed) into buf ----
  auto fill_tile = [&](int j, int buf) {
    const float* kp = Kg + (size_t)(j * 64) * D_;
    const float* vp = Vg + (size_t)(j * 64) * D_;
    unsigned short* dK = sK + buf * KV_TILE;
    unsigned int*   dV = (unsigned int*)(sVt + buf * KV_TILE);  // dword view, stride 36
    // K: row-major, 64 rows x 16 float4 segs, packed pairs -> b64 stores
#pragma unroll
    for (int it = 0; it < 4; ++it) {
      int row = frow + it * 16;
      float4 kf = *(const float4*)(kp + row * 64 + fseg * 4);
      uint2 kpkt;
      kpkt.x = packbf(kf.x, kf.y);
      kpkt.y = packbf(kf.z, kf.w);
      *(uint2*)&dK[row * KV_STRIDE + fseg * 4] = kpkt;
    }
    // V: transposed store, key PAIRS packed into one b32 per d
#pragma unroll
    for (int it = 0; it < 2; ++it) {
      int rp = frp + it * 16;
      const float* v0 = vp + (2 * rp) * 64 + fseg * 4;
      float4 va = *(const float4*)v0;
      float4 vb = *(const float4*)(v0 + 64);
      dV[(fseg * 4 + 0) * 36 + rp] = packbf(va.x, vb.x);
      dV[(fseg * 4 + 1) * 36 + rp] = packbf(va.y, vb.y);
      dV[(fseg * 4 + 2) * 36 + rp] = packbf(va.z, vb.z);
      dV[(fseg * 4 + 3) * 36 + rp] = packbf(va.w, vb.w);
    }
  };

  // ---- preload this wave's Q^T B-operands (pre-scaled by 1/sqrt(D)) ----
  // B as packed here (32x16, contiguous k): lane n<16: VGPR g holds K=2g,2g+1 ; lane n+16: K=16+2g,17+2g
  // => lane holds 16 contiguous d-values starting at kk*32 + sel*16 of its Q row.
  BF16x16 qb[2];
  {
    const float* qp = Qg + (size_t)qg * D_;
#pragma unroll
    for (int kk = 0; kk < 2; ++kk) {
      const float* p = qp + kk * 32 + sel * 16;
#pragma unroll
      for (int g = 0; g < 8; ++g)
        qb[kk].u[g] = packbf(p[2 * g] * 0.125f, p[2 * g + 1] * 0.125f);
    }
  }

  float mrow = -1e30f;   // running max (per q column == per lane)
  float lrow = 0.0f;     // running sum
  v8f o[4] = {};         // O^T accum: tile t -> d = 16t + r + 8*sel, q = qc

  fill_tile(0, 0);       // prologue: stage first tile

  for (int j = 0; j <= jmax; ++j) {
    // One barrier per key block: makes buf[j&1] visible AND orders last
    // iteration's reads of buf[(j+1)&1] before we overwrite it below.
    __syncthreads();
    if (j < jmax) fill_tile(j + 1, (j + 1) & 1);   // overlaps compute below

    const unsigned short* bK = sK  + (j & 1) * KV_TILE;
    const unsigned short* bV = sVt + (j & 1) * KV_TILE;
    const bool diag = (j >= jmax - 1);  // block-uniform: last two key blocks

    // ---- S^T tiles: t -> keys 16t..16t+15 (rows), q columns across lanes ----
    // C layout: VGPR r, lane -> (key_local = 16t + r + 8*sel, q = qc)
    v8f s[4];
#pragma unroll
    for (int t = 0; t < 4; ++t) {
      v8f acc = {};
#pragma unroll
      for (int kk = 0; kk < 2; ++kk) {
        // A layout (16x32 bf16): lane m row = 16t + qc;
        // halves 0-15 = d kk*32+16*sel .. +15 (contiguous)
        BF16x16 a;
        const unsigned short* ap = &bK[(16 * t + qc) * KV_STRIDE + kk * 32 + sel * 16];
        a.q[0] = *(const uint4*)ap;
        a.q[1] = *(const uint4*)(ap + 8);
        acc = __builtin_amdgcn_wmma_f32_16x16x32_bf16(
            false, a.v, false, qb[kk].v, (short)0, acc, false, false);
      }
      s[t] = acc;
    }

    // ---- causal mask (diagonal blocks only) + online softmax stats ----
    float mcur = -1e30f;
#pragma unroll
    for (int t = 0; t < 4; ++t)
#pragma unroll
      for (int r = 0; r < 8; ++r) {
        float x = s[t][r];
        if (diag) {
          int kg = j * 64 + 16 * t + r + 8 * sel;  // global key index
          x = (kg > qg) ? -1e30f : x;
          s[t][r] = x;
        }
        mcur = fmaxf(mcur, x);
      }
    {  // combine with other half (holds keys+8): one permlane16_swap + fmax
      U2 mm = halfswap(__float_as_uint(mcur), __float_as_uint(mcur));
      mcur = fmaxf(__uint_as_float(mm.x), __uint_as_float(mm.y));
    }
    const float mnew = fmaxf(mrow, mcur);
    const float mn2  = mnew * 1.44269504f;             // exp2-domain max
    const float corr = fast_exp2(__builtin_fmaf(mrow, 1.44269504f, -mn2));
    mrow = mnew;

    float psum = 0.0f;
#pragma unroll
    for (int t = 0; t < 4; ++t)
#pragma unroll
      for (int r = 0; r < 8; ++r) {
        // p = 2^(s*log2e - m*log2e): one FMA + one v_exp_f32 (masked -> 0)
        float p = fast_exp2(__builtin_fmaf(s[t][r], 1.44269504f, -mn2));
        s[t][r] = p;
        psum += p;
      }
    {
      U2 ss = halfswap(__float_as_uint(psum), __float_as_uint(psum));
      psum = __uint_as_float(ss.x) + __uint_as_float(ss.y);
    }
    lrow = lrow * corr + psum;

#pragma unroll
    for (int t = 0; t < 4; ++t)
#pragma unroll
      for (int r = 0; r < 8; ++r) o[t][r] *= corr;

    // ---- O^T += V^T * P^T ----
#pragma unroll
    for (int kk = 0; kk < 2; ++kk) {
      // Build B operand (32 keys x 16 q) from C-layout P^T tiles s[2kk], s[2kk+1].
      // permlane16_swap(p0, p1) gives exactly:
      //   res.x: lane<16 -> p0(self, keys 2h..), lane>=16 -> p1 from lane-16 (keys 16+2h)
      //   res.y: lane<16 -> p0 from lane+16 (keys 8+2h), lane>=16 -> p1(self, keys 24+2h)
      BF16x16 b;
#pragma unroll
      for (int h = 0; h < 4; ++h) {
        unsigned int p0 = packbf(s[2 * kk][2 * h], s[2 * kk][2 * h + 1]);
        unsigned int p1 = packbf(s[2 * kk + 1][2 * h], s[2 * kk + 1][2 * h + 1]);
        U2 r = halfswap(p0, p1);
        b.u[h]     = r.x;
        b.u[4 + h] = r.y;
      }
#pragma unroll
      for (int t = 0; t < 4; ++t) {
        // A = V^T rows: lane row d = 16t + qc; keys kk*32 + 16*sel .. +15
        BF16x16 a;
        const unsigned short* ap = &bV[(16 * t + qc) * KV_STRIDE + kk * 32 + sel * 16];
        a.q[0] = *(const uint4*)ap;
        a.q[1] = *(const uint4*)(ap + 8);
        o[t] = __builtin_amdgcn_wmma_f32_16x16x32_bf16(
            false, a.v, false, b.v, (short)0, o[t], false, false);
      }
    }
  }

  // ---- epilogue: O = O^T / l, transpose via LDS, coalesced float4 stores ----
  const float inv = fast_rcp(lrow);
  float* ob = sO + w * (16 * 68);
#pragma unroll
  for (int t = 0; t < 4; ++t)
#pragma unroll
    for (int r = 0; r < 8; ++r)
      ob[qc * 68 + 16 * t + r + 8 * sel] = o[t][r] * inv;
  // same-wave LDS ops are in-order; no cross-wave sharing of this region
#pragma unroll
  for (int it = 0; it < 8; ++it) {
    int idx = it * 32 + lane;
    int row = idx >> 4, seg = idx & 15;
    float4 val = *(const float4*)(ob + row * 68 + seg * 4);
    *(float4*)(Og + (size_t)(q0 + w * 16 + row) * D_ + seg * 4) = val;
  }
}

extern "C" void kernel_launch(void* const* d_in, const int* in_sizes, int n_in,
                              void* d_out, int out_size, void* d_ws, size_t ws_size,
                              hipStream_t stream) {
  (void)in_sizes; (void)n_in; (void)out_size; (void)d_ws; (void)ws_size;
  const float* Q = (const float*)d_in[0];
  const float* K = (const float*)d_in[1];
  const float* V = (const float*)d_in[2];
  float* O = (float*)d_out;
  dim3 grid(S_ / 128, B_ * H_);   // 16 query blocks x 32 (b,h)
  fa_cdna5_kernel<<<grid, 256, 0, stream>>>(Q, K, V, O);
}
